// FakeLinear_84834194030976
// MI455X (gfx1250) — matrix operand order repaired and checked by reading, not compiled, not run on hardware
//
#include <hip/hip_runtime.h>

// ---------------------------------------------------------------- types ----
typedef __bf16        v16bf __attribute__((ext_vector_type(16)));
typedef float         v8f   __attribute__((ext_vector_type(8)));
typedef unsigned int  v4u   __attribute__((ext_vector_type(4)));
typedef int           v8i   __attribute__((ext_vector_type(8)));
typedef int           v4i   __attribute__((ext_vector_type(4)));

// ------------------------------------------------------------- problem -----
#define MDIM 8192
#define NDIM 4096
#define KDIM 4096
#define BM   128
#define BN   128
#define BK   32
#define KT   (KDIM / BK)      // 128 k-steps
#define ROWB 80               // 64B of bf16 data + 16B pad (TDM pad_enable)
#define TILEB (BM * ROWB)     // 10240 B per tile
#define BUFB  (3 * TILEB)     // A + Wdir + Wrad per buffer
// -------------------------------------------------------------------------

static __device__ __forceinline__ float softplus_f(float x) {
    return x > 20.f ? x : log1pf(__expf(x));
}

static __device__ __forceinline__ unsigned short f2bf(float x) {
    union { float f; unsigned int u; } a; a.f = x;
    unsigned int r = a.u + 0x7fffu + ((a.u >> 16) & 1u);   // round-nearest-even
    return (unsigned short)(r >> 16);
}

// fp32 -> bf16 pack, 4 elements per thread per trip
__global__ void cvt_f32_bf16(const float* __restrict__ src,
                             unsigned short* __restrict__ dst, long long n4) {
    long long i = (long long)blockIdx.x * blockDim.x + threadIdx.x;
    long long stride = (long long)gridDim.x * blockDim.x;
    for (; i < n4; i += stride) {
        float4 f = reinterpret_cast<const float4*>(src)[i];
        union { unsigned short u[4]; unsigned long long v; } o;
        o.u[0] = f2bf(f.x); o.u[1] = f2bf(f.y);
        o.u[2] = f2bf(f.z); o.u[3] = f2bf(f.w);
        reinterpret_cast<unsigned long long*>(dst)[i] = o.v;
    }
}

// ---- TDM: load a [rows x cols] bf16 tile (row stride = strideElems) into
// ---- LDS at ldsByteAddr, padding each 64B row with 16B (ROWB = 80 stride).
static __device__ __forceinline__ void tdm_load_tile(const unsigned short* gsrc,
                                                     unsigned int ldsByteAddr,
                                                     unsigned int rows,
                                                     unsigned int cols,
                                                     unsigned int strideElems) {
    unsigned long long ga = (unsigned long long)gsrc;
    v4u g0;
    g0[0] = 1u;                                            // count = 1 (valid D#)
    g0[1] = ldsByteAddr;                                   // lds_addr
    g0[2] = (unsigned int)(ga & 0xffffffffull);            // global_addr[31:0]
    g0[3] = (unsigned int)((ga >> 32) & 0x1ffffffull)      // global_addr[56:32]
          | (2u << 30);                                    // type = 2 (image)
    v8i g1;
    unsigned int td0 = cols, td1 = rows;
    unsigned long long st0 = strideElems;
    g1[0] = (int)((1u << 16)       // data_size = 2 bytes
                | (1u << 20)       // pad_enable
                | (3u << 22)       // pad_interval: every 16 DW (64 B)
                | (3u << 25));     // pad_amount: 4 DW (16 B)
    g1[1] = (int)((td0 & 0xffffu) << 16);                  // tensor_dim0[15:0]
    g1[2] = (int)((td0 >> 16) | ((td1 & 0xffffu) << 16));  // td0[31:16] | td1[15:0]
    g1[3] = (int)((td1 >> 16) | ((cols & 0xffffu) << 16)); // td1[31:16] | tile_dim0
    g1[4] = (int)(rows & 0xffffu);                         // tile_dim1 (tile_dim2 = 0)
    g1[5] = (int)(st0 & 0xffffffffull);                    // tensor_dim0_stride lo
    g1[6] = (int)((st0 >> 32) & 0xffffull);                // stride hi | dim1_stride lo
    g1[7] = 0;
    v4i z4 = {0, 0, 0, 0};
    v8i z8 = {0, 0, 0, 0, 0, 0, 0, 0};
    // clang-23 / therock-10.0 6-arg form: (g0, g1, g2, g3, g4, cpol)
    __builtin_amdgcn_tensor_load_to_lds(g0, g1, z4, z4, z8, 0);
}

// A fragment (16x32 bf16, M x K): lanes 0-15 -> K{0..7,16..23}, 16-31 -> K{8..15,24..31}
static __device__ __forceinline__ v16bf load_fragA(const unsigned char* row, int half) {
    union { v16bf v; v4u q[2]; } u;
    u.q[0] = *(const v4u*)(row + half * 16);
    u.q[1] = *(const v4u*)(row + 32 + half * 16);
    return u.v;
}
// B fragment (32x16 bf16, K x N) built from row-major W[n,k]:
// lanes 0-15 -> K{0..7,16..23}, lanes 16-31 -> K{8..15,24..31}
static __device__ __forceinline__ v16bf load_fragB(const unsigned char* row, int half) {
    union { v16bf v; v4u q[2]; } u;
    u.q[0] = *(const v4u*)(row + half * 16);
    u.q[1] = *(const v4u*)(row + 32 + half * 16);
    return u.v;
}

__global__ __launch_bounds__(256) void fused_dual_gemm(
        const unsigned short* __restrict__ Abf,   // [MDIM x KDIM] bf16
        const unsigned short* __restrict__ Dbf,   // [NDIM x KDIM] bf16 (dir_loc)
        const unsigned short* __restrict__ Rbf,   // [NDIM x KDIM] bf16 (rad_w)
        const float* __restrict__ craw,           // [NDIM] dir_conc_raw
        const float* __restrict__ radb,           // [NDIM]
        const float* __restrict__ bias,           // [NDIM]
        float* __restrict__ out) {                // [MDIM x NDIM]
    __shared__ __align__(16) unsigned char smem[2 * BUFB];

    const int lane = threadIdx.x & 31;
    const int wave = threadIdx.x >> 5;   // 0..7
    const int half = lane >> 4;          // 0/1
    const int r    = lane & 15;
    const int wm   = wave >> 1;          // 0..3 -> 32-row strip
    const int wn   = wave & 1;           // 0..1 -> 64-col strip
    const int m0   = blockIdx.y * BM;
    const int n0   = blockIdx.x * BN;

    v8f accD[2][4], accR[2][4];
    const v8f vzero = {0.f, 0.f, 0.f, 0.f, 0.f, 0.f, 0.f, 0.f};
#pragma unroll
    for (int i = 0; i < 2; ++i)
#pragma unroll
        for (int j = 0; j < 4; ++j) { accD[i][j] = vzero; accR[i][j] = vzero; }

    // flat-LDS low 32 bits == LDS byte offset
    const unsigned int smemBase =
        (unsigned int)(unsigned long long)(void*)&smem[0];

    auto issue = [&](int kt, int buf) {
        const unsigned long long col = (unsigned long long)kt * BK;
        const unsigned int base = smemBase + (unsigned int)buf * BUFB;
        tdm_load_tile(Abf + (unsigned long long)m0 * KDIM + col, base,             BM, BK, KDIM);
        tdm_load_tile(Dbf + (unsigned long long)n0 * KDIM + col, base + TILEB,     BN, BK, KDIM);
        tdm_load_tile(Rbf + (unsigned long long)n0 * KDIM + col, base + 2 * TILEB, BN, BK, KDIM);
    };

    if (wave == 0) issue(0, 0);

    for (int kt = 0; kt < KT; ++kt) {
        const int buf = kt & 1;
        if (wave == 0) {
            if (kt + 1 < KT) {
                issue(kt + 1, buf ^ 1);
                __builtin_amdgcn_s_wait_tensorcnt(3);  // tiles for kt are done
            } else {
                __builtin_amdgcn_s_wait_tensorcnt(0);
            }
        }
        __syncthreads();  // LDS tiles for kt visible to all waves

        const unsigned char* As = smem + buf * BUFB;
        const unsigned char* Ds = As + TILEB;
        const unsigned char* Rs = As + 2 * TILEB;

        v16bf aF[2];
#pragma unroll
        for (int i = 0; i < 2; ++i)
            aF[i] = load_fragA(As + (wm * 32 + i * 16 + r) * ROWB, half);

#pragma unroll
        for (int j = 0; j < 4; ++j) {
            const int wrow = (wn * 64 + j * 16 + r) * ROWB;
            v16bf bD = load_fragB(Ds + wrow, half);
            v16bf bR = load_fragB(Rs + wrow, half);
#pragma unroll
            for (int i = 0; i < 2; ++i) {
                accD[i][j] = __builtin_amdgcn_wmma_f32_16x16x32_bf16(
                    false, aF[i], false, bD, (short)0, accD[i][j], false, false);
                accR[i][j] = __builtin_amdgcn_wmma_f32_16x16x32_bf16(
                    false, aF[i], false, bR, (short)0, accR[i][j], false, false);
            }
        }
        __syncthreads();  // compute done before buf is overwritten
    }

    // epilogue: out = accD * scale[n] * softplus(accR + rad_b[n]) + bias[n]
#pragma unroll
    for (int j = 0; j < 4; ++j) {
        const int n = n0 + wn * 64 + j * 16 + r;
        const float kappa = softplus_f(craw[n]);
        const float scale = kappa / (kappa + (float)(KDIM - 1));
        const float rb = radb[n];
        const float bs = bias[n];
#pragma unroll
        for (int i = 0; i < 2; ++i) {
            const int mb = m0 + wm * 32 + i * 16 + half * 8;
#pragma unroll
            for (int v = 0; v < 8; ++v) {
                const float rad = softplus_f(accR[i][j][v] + rb);
                out[(long long)(mb + v) * NDIM + n] = accD[i][j][v] * scale * rad + bs;
            }
        }
    }
}

extern "C" void kernel_launch(void* const* d_in, const int* in_sizes, int n_in,
                              void* d_out, int out_size, void* d_ws, size_t ws_size,
                              hipStream_t stream) {
    (void)in_sizes; (void)n_in; (void)out_size; (void)ws_size;
    const float* inp  = (const float*)d_in[0];  // [8192,4096]
    const float* dirL = (const float*)d_in[1];  // [4096,4096]
    const float* crw  = (const float*)d_in[2];  // [4096]
    const float* radw = (const float*)d_in[3];  // [4096,4096]
    const float* radb = (const float*)d_in[4];  // [4096]
    const float* bias = (const float*)d_in[5];  // [4096]
    float* out = (float*)d_out;

    unsigned short* Abf = (unsigned short*)d_ws;                 // 64 MB
    unsigned short* Dbf = Abf + (size_t)MDIM * KDIM;             // 32 MB
    unsigned short* Rbf = Dbf + (size_t)NDIM * KDIM;             // 32 MB

    const long long nA4 = (long long)MDIM * KDIM / 4;
    const long long nW4 = (long long)NDIM * KDIM / 4;
    cvt_f32_bf16<<<4096, 256, 0, stream>>>(inp,  Abf, nA4);
    cvt_f32_bf16<<<2048, 256, 0, stream>>>(dirL, Dbf, nW4);
    cvt_f32_bf16<<<2048, 256, 0, stream>>>(radw, Rbf, nW4);

    dim3 grid(NDIM / BN, MDIM / BM);  // (32, 64) = 2048 workgroups
    fused_dual_gemm<<<grid, 256, 0, stream>>>(Abf, Dbf, Rbf, crw, radb, bias, out);
}
